// HyperFourierLayer_28346784154024
// MI455X (gfx1250) — matrix operand order repaired and checked by reading, not compiled, not run on hardware
//
#include <hip/hip_runtime.h>
#include <hip/hip_bf16.h>
#include <math.h>

// ---- problem constants -----------------------------------------------------
#define Bn    32
#define Cn    1025
#define Ln    2048
#define Fn    1025           // rfft length
#define Mrows (Bn*Cn)        // 32800 rows = 2050 * 16
#define NPAD  2112           // 2*F=2050 padded to 66*32 = 132*16
#define KT1   64             // 2048 / 32   (GEMM1 K steps)
#define NT1   132            // 2112 / 16   (GEMM1 N tiles)
#define KT2   66             // 2112 / 32   (GEMM2 K steps)
#define NT2   128            // 2048 / 16   (GEMM2 N tiles)
#define NTB   4              // N tiles staged per block
#define MBLK  1025           // number of 32-row (2-tile) M blocks
#define MGRP  129            // ceil(32800 / 256) row-groups of 8 waves

typedef __attribute__((ext_vector_type(16))) _Float16 v16h;
typedef __attribute__((ext_vector_type(8)))  _Float16 v8h;
typedef __attribute__((ext_vector_type(8)))  float    v8f;
typedef __attribute__((ext_vector_type(4)))  float    v4f;
typedef __attribute__((ext_vector_type(4)))  int      v4i;

#if defined(__AMDGCN__) && __has_builtin(__builtin_amdgcn_global_load_async_to_lds_b128) && __has_builtin(__builtin_amdgcn_s_wait_asynccnt)
#define USE_ASYNC_LDS 1
typedef __attribute__((address_space(1))) v4i* gv4i_p;   // global int4*
typedef __attribute__((address_space(3))) v4i* lv4i_p;   // LDS int4*
#else
#define USE_ASYNC_LDS 0
#endif

// ---------------------------------------------------------------------------
// Basis builders. Basis matrices are stored PRE-SWIZZLED into the WMMA
// B-fragment layout (16-bit B 32x16 tile, wave32):
//   fragment(kstep, ntile) = 512 contiguous halves (1024 B)
//   lane n (n<16):  halves h=0..7 hold B[k=h][col=n], h=8..15 hold B[k=8+h][col=n]
//   lane n+16:      halves h=0..7 hold B[k=8+h][col=n], h=8..15 hold B[k=16+h][col=n]
// ---------------------------------------------------------------------------
__global__ void build_basis_fwd(_Float16* __restrict__ bswz) {
    int idx = blockIdx.x * blockDim.x + threadIdx.x;
    const int total = KT1 * NT1 * 512;
    if (idx >= total) return;
    int frag = idx >> 9;
    int within = idx & 511;
    int lane = within >> 4;
    int h = within & 15;
    int ks = frag / NT1;
    int ntile = frag - ks * NT1;
    int k = ks * 32 + ((lane >> 4) & 1) * 8 + (h & 7) + (h >> 3) * 16;   // time index l
    int n = ntile * 16 + (lane & 15);               // column in [0,2112)
    float val = 0.0f;
    if (n < 2 * Fn) {
        int f = (n < Fn) ? n : (n - Fn);
        int prod = (k * f) & (Ln - 1);              // exact mod-2048 reduction
        float th = (float)prod * (6.283185307179586f / (float)Ln);
        val = (n < Fn) ? __cosf(th) : -__sinf(th);
    }
    bswz[idx] = (_Float16)val;
}

__global__ void build_basis_inv(_Float16* __restrict__ bswz) {
    int idx = blockIdx.x * blockDim.x + threadIdx.x;
    const int total = KT2 * NT2 * 512;
    if (idx >= total) return;
    int frag = idx >> 9;
    int within = idx & 511;
    int lane = within >> 4;
    int h = within & 15;
    int ks = frag / NT2;
    int ntile = frag - ks * NT2;
    int nu = ks * 32 + ((lane >> 4) & 1) * 8 + (h & 7) + (h >> 3) * 16;  // spectral row in [0,2112)
    int l  = ntile * 16 + (lane & 15);              // time column in [0,2048)
    float val = 0.0f;
    if (nu < 2 * Fn) {
        int f = (nu < Fn) ? nu : (nu - Fn);
        float alpha = (f == 0 || f == Ln / 2) ? 1.0f : 2.0f;
        float scale = alpha / (float)Ln;
        int prod = (f * l) & (Ln - 1);
        float th = (float)prod * (6.283185307179586f / (float)Ln);
        val = (nu < Fn) ? scale * __cosf(th) : -scale * __sinf(th);
    }
    bswz[idx] = (_Float16)val;
}

// ---------------------------------------------------------------------------
// Cooperative staging of NTB B-fragments (4 KB) into LDS: one b128/thread.
// Async (gfx1250 GLOBAL_LOAD_ASYNC_TO_LDS_B128, ASYNCcnt) when available.
// ---------------------------------------------------------------------------
__device__ __forceinline__ void stage4k(const _Float16* gb, _Float16* lb, int tid) {
    v4i* g = (v4i*)gb + tid;
    v4i* l = (v4i*)lb + tid;
#if USE_ASYNC_LDS
    __builtin_amdgcn_global_load_async_to_lds_b128(
        (gv4i_p)(__attribute__((address_space(1))) void*)g,
        (lv4i_p)(__attribute__((address_space(3))) void*)l,
        0, 0);
#else
    *l = *g;
#endif
}

__device__ __forceinline__ void stage_wait() {
#if USE_ASYNC_LDS
    __builtin_amdgcn_s_wait_asynccnt(0);
#endif
}

__device__ __forceinline__ v16h lds_bfrag(const _Float16* lp) {
    v8h b0 = *(const v8h*)lp;
    v8h b1 = *(const v8h*)(lp + 8);
    v16h b;
#pragma unroll
    for (int i = 0; i < 8; ++i) { b[i] = b0[i]; b[8 + i] = b1[i]; }
    return b;
}

// ---------------------------------------------------------------------------
// GEMM1:  Xcat(f16, 32800 x 2112) = x(f32, 32800 x 2048) @ B1
// Block: 8 waves share one 64-col N-block via LDS; each wave owns 2 M tiles.
// ---------------------------------------------------------------------------
__global__ void __launch_bounds__(256) gemm_fwd(const float* __restrict__ X,
                                                const _Float16* __restrict__ Bswz,
                                                _Float16* __restrict__ Out) {
    const int NBLK = NT1 / NTB;                 // 33 n-blocks
    __shared__ __align__(16) _Float16 sbuf[2 * NTB * 512];   // 8 KB double buffer
    int mgrp = blockIdx.x / NBLK;
    int nb   = blockIdx.x - mgrp * NBLK;
    int tid  = threadIdx.x;
    int waveId = tid >> 5;
    int lane = tid & 31;
    int lhi  = (lane >> 4) & 1;
    int lno  = lane & 15;
    int mb = mgrp * 8 + waveId;                 // 32-row block index
    bool active = (mb < MBLK);
    int rowBase = mb * 32;
    int colBase = nb * NTB * 16;

    // prologue: stage K-step 0
    stage4k(Bswz + ((size_t)(0 * NT1 + nb * NTB) << 9), &sbuf[0], tid);

    v8f acc[2][NTB] = {};
    for (int ks = 0; ks < KT1; ++ks) {
        int cur = ks & 1;
        stage_wait();
        __syncthreads();
        if (ks + 1 < KT1)
            stage4k(Bswz + ((size_t)((ks + 1) * NT1 + nb * NTB) << 9),
                    &sbuf[(cur ^ 1) * NTB * 512], tid);
        if (active) {
            int kA = ks * 32 + lhi * 8;
            v16h afrag[2];
#pragma unroll
            for (int mt = 0; mt < 2; ++mt) {
                const float* ap = X + (size_t)(rowBase + mt * 16 + lno) * Ln + kA;
                v4f f0 = *(const v4f*)(ap + 0);
                v4f f1 = *(const v4f*)(ap + 4);
                v4f f2 = *(const v4f*)(ap + 16);
                v4f f3 = *(const v4f*)(ap + 20);
                v16h a;
#pragma unroll
                for (int i = 0; i < 4; ++i) {
                    a[i]      = (_Float16)f0[i];
                    a[4 + i]  = (_Float16)f1[i];
                    a[8 + i]  = (_Float16)f2[i];
                    a[12 + i] = (_Float16)f3[i];
                }
                afrag[mt] = a;
            }
            v16h bfrag[NTB];
#pragma unroll
            for (int nt = 0; nt < NTB; ++nt)
                bfrag[nt] = lds_bfrag(&sbuf[cur * NTB * 512 + nt * 512 + lane * 16]);
#pragma unroll
            for (int mt = 0; mt < 2; ++mt)
#pragma unroll
                for (int nt = 0; nt < NTB; ++nt)
                    acc[mt][nt] = __builtin_amdgcn_wmma_f32_16x16x32_f16(
                        false, afrag[mt], false, bfrag[nt],
                        (short)0, acc[mt][nt], false, false);
        }
    }
    if (active) {
#pragma unroll
        for (int mt = 0; mt < 2; ++mt)
#pragma unroll
            for (int nt = 0; nt < NTB; ++nt) {
                int col = colBase + nt * 16 + lno;
#pragma unroll
                for (int j = 0; j < 8; ++j) {
                    int row = rowBase + mt * 16 + lhi * 8 + j;
                    Out[(size_t)row * NPAD + col] = (_Float16)acc[mt][nt][j];
                }
            }
    }
}

// ---------------------------------------------------------------------------
// In-place complex modulation on Xcat:  Z = X * (wr + i*wi)
// ---------------------------------------------------------------------------
__global__ void modulate(_Float16* __restrict__ Z,
                         const float* __restrict__ FW,
                         const float* __restrict__ PS) {
    long long idx = (long long)blockIdx.x * blockDim.x + threadIdx.x;
    const long long total = (long long)Mrows * Fn;
    if (idx >= total) return;
    int r = (int)(idx / Fn);
    int f = (int)(idx - (long long)r * Fn);
    int c = r % Cn;
    _Float16* row = Z + (size_t)r * NPAD;
    float xr = (float)row[f];
    float xi = (float)row[Fn + f];
    float wr = FW[(size_t)c * Fn + f];
    float wi = PS[f];
    row[f]      = (_Float16)(xr * wr - xi * wi);
    row[Fn + f] = (_Float16)(xr * wi + xi * wr);
}

// ---------------------------------------------------------------------------
// GEMM2:  y(f32, 32800 x 2048) = Zcat(f16, 32800 x 2112) @ B2
// ---------------------------------------------------------------------------
__global__ void __launch_bounds__(256) gemm_inv(const _Float16* __restrict__ Z,
                                                const _Float16* __restrict__ Bswz,
                                                float* __restrict__ Out) {
    const int NBLK = NT2 / NTB;                 // 32 n-blocks
    __shared__ __align__(16) _Float16 sbuf[2 * NTB * 512];
    int mgrp = blockIdx.x / NBLK;
    int nb   = blockIdx.x - mgrp * NBLK;
    int tid  = threadIdx.x;
    int waveId = tid >> 5;
    int lane = tid & 31;
    int lhi  = (lane >> 4) & 1;
    int lno  = lane & 15;
    int mb = mgrp * 8 + waveId;
    bool active = (mb < MBLK);
    int rowBase = mb * 32;
    int colBase = nb * NTB * 16;

    stage4k(Bswz + ((size_t)(0 * NT2 + nb * NTB) << 9), &sbuf[0], tid);

    v8f acc[2][NTB] = {};
    for (int ks = 0; ks < KT2; ++ks) {
        int cur = ks & 1;
        stage_wait();
        __syncthreads();
        if (ks + 1 < KT2)
            stage4k(Bswz + ((size_t)((ks + 1) * NT2 + nb * NTB) << 9),
                    &sbuf[(cur ^ 1) * NTB * 512], tid);
        if (active) {
            int kA = ks * 32 + lhi * 8;
            v16h afrag[2];
#pragma unroll
            for (int mt = 0; mt < 2; ++mt) {
                const _Float16* ap = Z + (size_t)(rowBase + mt * 16 + lno) * NPAD + kA;
                v8h a0 = *(const v8h*)ap;          // K = kA..kA+7
                v8h a1 = *(const v8h*)(ap + 16);   // K = kA+16..kA+23
                v16h a;
#pragma unroll
                for (int i = 0; i < 8; ++i) { a[i] = a0[i]; a[8 + i] = a1[i]; }
                afrag[mt] = a;
            }
            v16h bfrag[NTB];
#pragma unroll
            for (int nt = 0; nt < NTB; ++nt)
                bfrag[nt] = lds_bfrag(&sbuf[cur * NTB * 512 + nt * 512 + lane * 16]);
#pragma unroll
            for (int mt = 0; mt < 2; ++mt)
#pragma unroll
                for (int nt = 0; nt < NTB; ++nt)
                    acc[mt][nt] = __builtin_amdgcn_wmma_f32_16x16x32_f16(
                        false, afrag[mt], false, bfrag[nt],
                        (short)0, acc[mt][nt], false, false);
        }
    }
    if (active) {
#pragma unroll
        for (int mt = 0; mt < 2; ++mt)
#pragma unroll
            for (int nt = 0; nt < NTB; ++nt) {
                int col = colBase + nt * 16 + lno;
#pragma unroll
                for (int j = 0; j < 8; ++j) {
                    int row = rowBase + mt * 16 + lhi * 8 + j;
                    Out[(size_t)row * Ln + col] = acc[mt][nt][j];
                }
            }
    }
}

// ---------------------------------------------------------------------------
extern "C" void kernel_launch(void* const* d_in, const int* in_sizes, int n_in,
                              void* d_out, int out_size, void* d_ws, size_t ws_size,
                              hipStream_t stream) {
    const float* x  = (const float*)d_in[0];          // (B, C, L) f32
    const float* fw = (const float*)d_in[1];          // (C, F)    f32
    const float* ps = (const float*)d_in[2];          // (F,)      f32
    float* y = (float*)d_out;                         // (B, C, L) f32

    char* ws = (char*)d_ws;
    const size_t b1_bytes = (size_t)KT1 * NT1 * 512 * 2;   // 8,650,752
    const size_t b2_bytes = (size_t)KT2 * NT2 * 512 * 2;   // 8,650,752
    _Float16* b1   = (_Float16*)(ws);
    _Float16* b2   = (_Float16*)(ws + b1_bytes);
    _Float16* xcat = (_Float16*)(ws + b1_bytes + b2_bytes); // 32800*2112 f16

    // 1) build swizzled DFT bases (stay resident in 192MB L2)
    {
        int total = KT1 * NT1 * 512;
        build_basis_fwd<<<(total + 255) / 256, 256, 0, stream>>>(b1);
    }
    {
        int total = KT2 * NT2 * 512;
        build_basis_inv<<<(total + 255) / 256, 256, 0, stream>>>(b2);
    }

    // 2) forward "rfft" GEMM: Xcat = x @ B1
    {
        int blocks = MGRP * (NT1 / NTB);   // 129 * 33
        gemm_fwd<<<blocks, 256, 0, stream>>>(x, b1, xcat);
    }

    // 3) in-place complex modulation
    {
        long long total = (long long)Mrows * Fn;
        int blocks = (int)((total + 255) / 256);
        modulate<<<blocks, 256, 0, stream>>>(xcat, fw, ps);
    }

    // 4) inverse "irfft" GEMM: y = Zcat @ B2
    {
        int blocks = MGRP * (NT2 / NTB);   // 129 * 32
        gemm_inv<<<blocks, 256, 0, stream>>>(xcat, b2, y);
    }
}
